// MoLFFN_60507499266802
// MI455X (gfx1250) — matrix operand order repaired and checked by reading, not compiled, not run on hardware
//
#include <hip/hip_runtime.h>
#include <hip/hip_bf16.h>
#include <cstdint>
#include <cstddef>

// Problem dims (fixed by the reference)
#define NTOK 2048   // B*L
#define DIM  2048   // D
#define NEXP 8      // E
#define NRANK 8     // RANK
#define DFF  5504
#define RPAD 96     // 8 shared + 64 expert LoRA cols, zero-padded to 3 K-steps of 32
#define LPAD 40     // LDS stride (bf16 elems) for 32-wide tiles: 32 + 8 pad (16B)

typedef __attribute__((ext_vector_type(16))) __bf16 v16bf;
typedef __attribute__((ext_vector_type(8)))  float  v8f;

static_assert(DIM % 32 == 0, "");
static_assert(DFF % 64 == 0, "");
static_assert(NTOK % 128 == 0, "");

// ---------------------------------------------------------------------------
// WMMA helpers (wave32, 16x16x32 bf16 -> f32)
// ---------------------------------------------------------------------------
__device__ __forceinline__ v8f wmma_bf16(v16bf a, v16bf b, v8f c) {
  return __builtin_amdgcn_wmma_f32_16x16x32_bf16(false, a, false, b, (short)0, c,
                                                 false, false);
}

// A tile staged row-major in LDS: 128 rows x 32 K, row stride LPAD bf16.
__device__ __forceinline__ v16bf load_A_frag(const __bf16* sA16) {
  const int lane = threadIdx.x & 31;
  const int half = lane >> 4, m = lane & 15;
  v16bf f;
  ((uint4*)&f)[0] = *(const uint4*)(sA16 + m * LPAD + half * 8);
  ((uint4*)&f)[1] = *(const uint4*)(sA16 + m * LPAD + 16 + half * 8);
  return f;
}

// B tile staged column-major in LDS: cols x 32 K, col stride LPAD bf16.
__device__ __forceinline__ v16bf load_B_frag(const __bf16* sB16) {
  const int lane = threadIdx.x & 31;
  const int half = lane >> 4, n = lane & 15;
  v16bf f;
  ((uint4*)&f)[0] = *(const uint4*)(sB16 + n * LPAD + half * 8);
  ((uint4*)&f)[1] = *(const uint4*)(sB16 + n * LPAD + 16 + half * 8);
  return f;
}

// ---------------------------------------------------------------------------
// Data movement (assumes blockDim.x == 256)
// ---------------------------------------------------------------------------
// Async copy of a 128x32 bf16 A tile straight into LDS (no VGPR transit).
// Each thread moves 32B via two GLOBAL_LOAD_ASYNC_TO_LDS_B128 (ASYNCcnt).
__device__ __forceinline__ void async_A(__bf16* sA, const __bf16* src, int lda,
                                        int k0) {
  const int t = threadIdx.x;
  const int row = t >> 1, half = t & 1;
  const void* g = (const void*)(src + (size_t)row * lda + k0 + half * 16);
  const unsigned lds = (unsigned)(uintptr_t)(void*)(sA + row * LPAD + half * 16);
  asm volatile(
      "global_load_async_to_lds_b128 %0, %1, off\n\t"
      "global_load_async_to_lds_b128 %0, %1, off offset:16"
      :
      : "v"(lds), "v"(g)
      : "memory");
}

__device__ __forceinline__ void wait_async() {
  asm volatile("s_wait_asynccnt 0" ::: "memory");
}

// B tile: 32 K-rows x 32 cols. Global fp32 load to regs (issued early), then
// convert + transpose-store into column-major bf16 LDS tile.
__device__ __forceinline__ float4 loadB_f32(const float* src, int ldb, int k0,
                                            int n0) {
  const int t = threadIdx.x;
  const int r = t >> 3, cg = (t & 7) << 2;
  return *(const float4*)(src + (size_t)(k0 + r) * ldb + n0 + cg);
}
__device__ __forceinline__ void storeB(__bf16* sB, float4 v) {
  const int t = threadIdx.x;
  const int r = t >> 3, cg = (t & 7) << 2;
  sB[(cg + 0) * LPAD + r] = (__bf16)v.x;
  sB[(cg + 1) * LPAD + r] = (__bf16)v.y;
  sB[(cg + 2) * LPAD + r] = (__bf16)v.z;
  sB[(cg + 3) * LPAD + r] = (__bf16)v.w;
}

// Same for bf16-source B tiles (LoRA tail weights / SAcat).
__device__ __forceinline__ uint2 loadB_bf16(const __bf16* src, int ldb, int k0,
                                            int n0) {
  const int t = threadIdx.x;
  const int r = t >> 3, cg = (t & 7) << 2;
  return *(const uint2*)(src + (size_t)(k0 + r) * ldb + n0 + cg);
}
__device__ __forceinline__ void storeB_bf16(__bf16* sB, uint2 d) {
  const int t = threadIdx.x;
  const int r = t >> 3, cg = (t & 7) << 2;
  union {
    uint2 u;
    __bf16 h[4];
  } x;
  x.u = d;
  sB[(cg + 0) * LPAD + r] = x.h[0];
  sB[(cg + 1) * LPAD + r] = x.h[1];
  sB[(cg + 2) * LPAD + r] = x.h[2];
  sB[(cg + 3) * LPAD + r] = x.h[3];
}

// ---------------------------------------------------------------------------
// Prep kernels: build concatenated LoRA operands in bf16
// ---------------------------------------------------------------------------
__global__ void prep_cols_kernel(const float* __restrict__ sharedA,
                                 const float* __restrict__ expertA, int Din,
                                 __bf16* __restrict__ dst) {
  const int idx = blockIdx.x * blockDim.x + threadIdx.x;
  if (idx >= Din * RPAD) return;
  const int d = idx / RPAD, j = idx % RPAD;
  float v = 0.f;
  if (j < NRANK) {
    v = sharedA[(size_t)d * NRANK + j];
  } else if (j < NRANK + NEXP * NRANK) {
    const int e = (j - NRANK) >> 3, r = (j - NRANK) & 7;
    v = expertA[((size_t)e * Din + d) * NRANK + r];
  }
  dst[idx] = (__bf16)v;
}

__global__ void prep_rows_kernel(const float* __restrict__ sharedB,
                                 const float* __restrict__ expertB, int Dout,
                                 __bf16* __restrict__ dst) {
  const int idx = blockIdx.x * blockDim.x + threadIdx.x;
  if (idx >= RPAD * Dout) return;
  const int r = idx / Dout, n = idx % Dout;
  float v = 0.f;
  if (r < NRANK) {
    v = sharedB[(size_t)r * Dout + n];
  } else if (r < NRANK + NEXP * NRANK) {
    const int e = (r - NRANK) >> 3, rr = (r - NRANK) & 7;
    v = expertB[((size_t)e * NRANK + rr) * Dout + n];
  }
  dst[idx] = (__bf16)v;
}

__global__ void cast_kernel(const float* __restrict__ x, __bf16* __restrict__ xb,
                            int n) {
  const int i = blockIdx.x * blockDim.x + threadIdx.x;
  if (i < n) xb[i] = (__bf16)x[i];
}

// ---------------------------------------------------------------------------
// Router: logits -> sigmoid -> top2(biased) -> normalized expert weights
// ---------------------------------------------------------------------------
__global__ __launch_bounds__(256) void router_kernel(
    const float* __restrict__ x, const float* __restrict__ Wr,
    const float* __restrict__ bias, float* __restrict__ wout) {
  const int n = blockIdx.x;
  const int e = threadIdx.x >> 5;
  const int lane = threadIdx.x & 31;
  const float* xr = x + (size_t)n * DIM;
  float p = 0.f;
  for (int d = lane; d < DIM; d += 32) p += xr[d] * Wr[(size_t)d * NEXP + e];
  for (int off = 16; off > 0; off >>= 1) p += __shfl_down(p, off, 32);
  __shared__ float sl[NEXP];
  if (lane == 0) sl[e] = p;
  __syncthreads();
  if (threadIdx.x == 0) {
    float sc[NEXP], bi[NEXP];
    for (int i = 0; i < NEXP; ++i) {
      sc[i] = 1.f / (1.f + __expf(-sl[i]));
      bi[i] = sc[i] + bias[i];
    }
    int i0 = 0;
    for (int i = 1; i < NEXP; ++i)
      if (bi[i] > bi[i0]) i0 = i;
    int i1 = (i0 == 0) ? 1 : 0;
    for (int i = 0; i < NEXP; ++i)
      if (i != i0 && bi[i] > bi[i1]) i1 = i;
    const float s0 = sc[i0], s1 = sc[i1];
    const float inv = 1.f / (s0 + s1 + 1e-8f);
    for (int i = 0; i < NEXP; ++i) wout[(size_t)n * NEXP + i] = 0.f;
    wout[(size_t)n * NEXP + i0] = s0 * inv;
    wout[(size_t)n * NEXP + i1] = s1 * inv;
  }
}

// ---------------------------------------------------------------------------
// Small WMMA GEMM (pipelined, BN=32): P(N x RPAD) = A(N x K) @ Bsa(K x RPAD)
// grid = (RPAD/32, NTOK/128), 256 threads
// ---------------------------------------------------------------------------
__global__ __launch_bounds__(256) void gemm_u_kernel(
    const __bf16* __restrict__ A, int lda, int K, const __bf16* __restrict__ Bsa,
    float* __restrict__ P) {
  __shared__ __align__(16) __bf16 sA[2][128 * LPAD];
  __shared__ __align__(16) __bf16 sB[2][32 * LPAD];
  const int n0 = blockIdx.x * 32;
  const int row0 = blockIdx.y * 128;
  const int w = threadIdx.x >> 5;
  const __bf16* Arow = A + (size_t)row0 * lda;
  v8f a0 = {}, a1 = {};
  async_A(sA[0], Arow, lda, 0);
  storeB_bf16(sB[0], loadB_bf16(Bsa, RPAD, 0, n0));
  wait_async();
  __syncthreads();
  const int KT = K / 32;
  for (int kt = 0; kt < KT; ++kt) {
    const int cur = kt & 1, nxt = cur ^ 1;
    uint2 bn;
    if (kt + 1 < KT) {
      async_A(sA[nxt], Arow, lda, (kt + 1) * 32);
      bn = loadB_bf16(Bsa, RPAD, (kt + 1) * 32, n0);
    }
    const v16bf a = load_A_frag(sA[cur] + w * 16 * LPAD);
    a0 = wmma_bf16(a, load_B_frag(sB[cur]), a0);
    a1 = wmma_bf16(a, load_B_frag(sB[cur] + 16 * LPAD), a1);
    if (kt + 1 < KT) storeB_bf16(sB[nxt], bn);
    wait_async();
    __syncthreads();
  }
  const int lane = threadIdx.x & 31;
  const int half = lane >> 4, nn = lane & 15;
  const size_t rbase = row0 + w * 16 + half * 8;
  for (int i = 0; i < 8; ++i) {
    P[(rbase + i) * RPAD + n0 + nn] = a0[i];
    P[(rbase + i) * RPAD + n0 + 16 + nn] = a1[i];
  }
}

__global__ void make_u_kernel(const float* __restrict__ P,
                              const float* __restrict__ w,
                              __bf16* __restrict__ u) {
  const int n = blockIdx.x;
  const int j = threadIdx.x;  // 96 threads
  const float scale = 1.0f / (float)NRANK;
  const float p = P[(size_t)n * RPAD + j];
  float v = 0.f;
  if (j < NRANK) v = scale * p;
  else if (j < NRANK + NEXP * NRANK)
    v = scale * w[(size_t)n * NEXP + ((j - NRANK) >> 3)] * p;
  u[(size_t)n * RPAD + j] = (__bf16)v;
}

// ---------------------------------------------------------------------------
// Fused gate/up GEMM (pipelined, BM=128, BN=64) + LoRA tail + silu(gate)*up
// grid = (DFF/64, NTOK/128), 256 threads; 8 WMMAs per K-step per wave.
// ---------------------------------------------------------------------------
__global__ __launch_bounds__(256) void gateup_kernel(
    const __bf16* __restrict__ xb, const float* __restrict__ Wg,
    const float* __restrict__ Wu, const __bf16* __restrict__ ug,
    const __bf16* __restrict__ uu, const __bf16* __restrict__ Vg,
    const __bf16* __restrict__ Vu, __bf16* __restrict__ hidden) {
  __shared__ __align__(16) __bf16 sA[2][128 * LPAD];
  __shared__ __align__(16) __bf16 sBg[2][64 * LPAD];
  __shared__ __align__(16) __bf16 sBu[2][64 * LPAD];
  const int n0 = blockIdx.x * 64;
  const int row0 = blockIdx.y * 128;
  const int w = threadIdx.x >> 5;
  const __bf16* Arow = xb + (size_t)row0 * DIM;
  v8f ag[4] = {{}, {}, {}, {}};
  v8f au[4] = {{}, {}, {}, {}};
  // prologue
  async_A(sA[0], Arow, DIM, 0);
  storeB(sBg[0], loadB_f32(Wg, DFF, 0, n0));
  storeB(sBg[0] + 32 * LPAD, loadB_f32(Wg, DFF, 0, n0 + 32));
  storeB(sBu[0], loadB_f32(Wu, DFF, 0, n0));
  storeB(sBu[0] + 32 * LPAD, loadB_f32(Wu, DFF, 0, n0 + 32));
  wait_async();
  __syncthreads();
  const int KT = DIM / 32;
  for (int kt = 0; kt < KT; ++kt) {
    const int cur = kt & 1, nxt = cur ^ 1;
    float4 bg0N, bg1N, bu0N, bu1N;
    if (kt + 1 < KT) {
      const int k1 = (kt + 1) * 32;
      async_A(sA[nxt], Arow, DIM, k1);
      bg0N = loadB_f32(Wg, DFF, k1, n0);
      bg1N = loadB_f32(Wg, DFF, k1, n0 + 32);
      bu0N = loadB_f32(Wu, DFF, k1, n0);
      bu1N = loadB_f32(Wu, DFF, k1, n0 + 32);
      if (kt + 2 < KT) {  // pull kt+2 weight rows toward L2
        const int kn = (kt + 2) * 32 + (threadIdx.x >> 3);
        const int nc = n0 + ((threadIdx.x & 1) << 5);
        __builtin_prefetch(Wg + (size_t)kn * DFF + nc, 0, 1);
        __builtin_prefetch(Wu + (size_t)kn * DFF + nc, 0, 1);
      }
    }
    const v16bf a = load_A_frag(sA[cur] + w * 16 * LPAD);
#pragma unroll
    for (int g = 0; g < 4; ++g) {
      ag[g] = wmma_bf16(a, load_B_frag(sBg[cur] + g * 16 * LPAD), ag[g]);
      au[g] = wmma_bf16(a, load_B_frag(sBu[cur] + g * 16 * LPAD), au[g]);
    }
    if (kt + 1 < KT) {
      storeB(sBg[nxt], bg0N);
      storeB(sBg[nxt] + 32 * LPAD, bg1N);
      storeB(sBu[nxt], bu0N);
      storeB(sBu[nxt] + 32 * LPAD, bu1N);
    }
    wait_async();
    __syncthreads();
  }
  // LoRA tails folded as extra K-steps (gate, then up)
  for (int t = 0; t < RPAD / 32; ++t) {
    async_A(sA[0], ug + (size_t)row0 * RPAD, RPAD, t * 32);
    storeB_bf16(sBg[0], loadB_bf16(Vg, DFF, t * 32, n0));
    storeB_bf16(sBg[0] + 32 * LPAD, loadB_bf16(Vg, DFF, t * 32, n0 + 32));
    wait_async();
    __syncthreads();
    const v16bf a = load_A_frag(sA[0] + w * 16 * LPAD);
#pragma unroll
    for (int g = 0; g < 4; ++g)
      ag[g] = wmma_bf16(a, load_B_frag(sBg[0] + g * 16 * LPAD), ag[g]);
    __syncthreads();
  }
  for (int t = 0; t < RPAD / 32; ++t) {
    async_A(sA[0], uu + (size_t)row0 * RPAD, RPAD, t * 32);
    storeB_bf16(sBu[0], loadB_bf16(Vu, DFF, t * 32, n0));
    storeB_bf16(sBu[0] + 32 * LPAD, loadB_bf16(Vu, DFF, t * 32, n0 + 32));
    wait_async();
    __syncthreads();
    const v16bf a = load_A_frag(sA[0] + w * 16 * LPAD);
#pragma unroll
    for (int g = 0; g < 4; ++g)
      au[g] = wmma_bf16(a, load_B_frag(sBu[0] + g * 16 * LPAD), au[g]);
    __syncthreads();
  }
  // epilogue: hidden = silu(gate) * up, bf16
  const int lane = threadIdx.x & 31;
  const int half = lane >> 4, nn = lane & 15;
  const size_t rbase = row0 + w * 16 + half * 8;
#pragma unroll
  for (int g = 0; g < 4; ++g) {
    const int col = n0 + g * 16 + nn;
#pragma unroll
    for (int i = 0; i < 8; ++i) {
      const float gv = ag[g][i], uv = au[g][i];
      hidden[(rbase + i) * DFF + col] = (__bf16)(gv / (1.f + __expf(-gv)) * uv);
    }
  }
}

// ---------------------------------------------------------------------------
// Down projection (pipelined, BM=128, BN=64) + LoRA tail -> out (f32)
// grid = (DIM/64, NTOK/128), 256 threads; 4 WMMAs per K-step per wave.
// ---------------------------------------------------------------------------
__global__ __launch_bounds__(256) void down_kernel(
    const __bf16* __restrict__ hidden, const float* __restrict__ Wd,
    const __bf16* __restrict__ ud, const __bf16* __restrict__ Vd,
    float* __restrict__ out) {
  __shared__ __align__(16) __bf16 sA[2][128 * LPAD];
  __shared__ __align__(16) __bf16 sB[2][64 * LPAD];
  const int n0 = blockIdx.x * 64;
  const int row0 = blockIdx.y * 128;
  const int w = threadIdx.x >> 5;
  const __bf16* Arow = hidden + (size_t)row0 * DFF;
  v8f acc[4] = {{}, {}, {}, {}};
  async_A(sA[0], Arow, DFF, 0);
  storeB(sB[0], loadB_f32(Wd, DIM, 0, n0));
  storeB(sB[0] + 32 * LPAD, loadB_f32(Wd, DIM, 0, n0 + 32));
  wait_async();
  __syncthreads();
  const int KT = DFF / 32;
  for (int kt = 0; kt < KT; ++kt) {
    const int cur = kt & 1, nxt = cur ^ 1;
    float4 b0N, b1N;
    if (kt + 1 < KT) {
      const int k1 = (kt + 1) * 32;
      async_A(sA[nxt], Arow, DFF, k1);
      b0N = loadB_f32(Wd, DIM, k1, n0);
      b1N = loadB_f32(Wd, DIM, k1, n0 + 32);
      if (kt + 2 < KT) {
        const int kn = (kt + 2) * 32 + (threadIdx.x >> 3);
        const int nc = n0 + ((threadIdx.x & 1) << 5);
        __builtin_prefetch(Wd + (size_t)kn * DIM + nc, 0, 1);
      }
    }
    const v16bf a = load_A_frag(sA[cur] + w * 16 * LPAD);
#pragma unroll
    for (int g = 0; g < 4; ++g)
      acc[g] = wmma_bf16(a, load_B_frag(sB[cur] + g * 16 * LPAD), acc[g]);
    if (kt + 1 < KT) {
      storeB(sB[nxt], b0N);
      storeB(sB[nxt] + 32 * LPAD, b1N);
    }
    wait_async();
    __syncthreads();
  }
  for (int t = 0; t < RPAD / 32; ++t) {
    async_A(sA[0], ud + (size_t)row0 * RPAD, RPAD, t * 32);
    storeB_bf16(sB[0], loadB_bf16(Vd, DIM, t * 32, n0));
    storeB_bf16(sB[0] + 32 * LPAD, loadB_bf16(Vd, DIM, t * 32, n0 + 32));
    wait_async();
    __syncthreads();
    const v16bf a = load_A_frag(sA[0] + w * 16 * LPAD);
#pragma unroll
    for (int g = 0; g < 4; ++g)
      acc[g] = wmma_bf16(a, load_B_frag(sB[0] + g * 16 * LPAD), acc[g]);
    __syncthreads();
  }
  const int lane = threadIdx.x & 31;
  const int half = lane >> 4, nn = lane & 15;
  const size_t rbase = row0 + w * 16 + half * 8;
#pragma unroll
  for (int g = 0; g < 4; ++g) {
    const int col = n0 + g * 16 + nn;
#pragma unroll
    for (int i = 0; i < 8; ++i) out[(rbase + i) * DIM + col] = acc[g][i];
  }
}

// ---------------------------------------------------------------------------
// Launch
// ---------------------------------------------------------------------------
extern "C" void kernel_launch(void* const* d_in, const int* in_sizes, int n_in,
                              void* d_out, int out_size, void* d_ws,
                              size_t ws_size, hipStream_t stream) {
  const float* x       = (const float*)d_in[0];
  const float* W_rt    = (const float*)d_in[1];
  const float* W_gate  = (const float*)d_in[2];
  const float* W_up    = (const float*)d_in[3];
  const float* W_down  = (const float*)d_in[4];
  const float* sA_gate = (const float*)d_in[5];
  const float* sB_gate = (const float*)d_in[6];
  const float* sA_up   = (const float*)d_in[7];
  const float* sB_up   = (const float*)d_in[8];
  const float* sA_down = (const float*)d_in[9];
  const float* sB_down = (const float*)d_in[10];
  const float* A_gate  = (const float*)d_in[11];
  const float* B_gate  = (const float*)d_in[12];
  const float* A_up    = (const float*)d_in[13];
  const float* B_up    = (const float*)d_in[14];
  const float* A_down  = (const float*)d_in[15];
  const float* B_down  = (const float*)d_in[16];
  const float* ebias   = (const float*)d_in[17];
  float* out = (float*)d_out;

  char* p = (char*)d_ws;
  auto alloc = [&](size_t bytes) -> void* {
    void* r = (void*)p;
    p += (bytes + 255) & ~(size_t)255;
    return r;
  };
  float*  w_rout = (float*)alloc((size_t)NTOK * NEXP * 4);
  __bf16* xb     = (__bf16*)alloc((size_t)NTOK * DIM * 2);
  __bf16* hidden = (__bf16*)alloc((size_t)NTOK * DFF * 2);
  float*  P      = (float*)alloc((size_t)NTOK * RPAD * 4);
  __bf16* ug     = (__bf16*)alloc((size_t)NTOK * RPAD * 2);
  __bf16* uu     = (__bf16*)alloc((size_t)NTOK * RPAD * 2);
  __bf16* ud     = (__bf16*)alloc((size_t)NTOK * RPAD * 2);
  __bf16* SAg    = (__bf16*)alloc((size_t)DIM * RPAD * 2);
  __bf16* SAu    = (__bf16*)alloc((size_t)DIM * RPAD * 2);
  __bf16* SAd    = (__bf16*)alloc((size_t)DFF * RPAD * 2);
  __bf16* Vg     = (__bf16*)alloc((size_t)RPAD * DFF * 2);
  __bf16* Vu     = (__bf16*)alloc((size_t)RPAD * DFF * 2);
  __bf16* Vd     = (__bf16*)alloc((size_t)RPAD * DIM * 2);

  prep_cols_kernel<<<(DIM * RPAD + 255) / 256, 256, 0, stream>>>(sA_gate, A_gate, DIM, SAg);
  prep_cols_kernel<<<(DIM * RPAD + 255) / 256, 256, 0, stream>>>(sA_up, A_up, DIM, SAu);
  prep_cols_kernel<<<(DFF * RPAD + 255) / 256, 256, 0, stream>>>(sA_down, A_down, DFF, SAd);
  prep_rows_kernel<<<(RPAD * DFF + 255) / 256, 256, 0, stream>>>(sB_gate, B_gate, DFF, Vg);
  prep_rows_kernel<<<(RPAD * DFF + 255) / 256, 256, 0, stream>>>(sB_up, B_up, DFF, Vu);
  prep_rows_kernel<<<(RPAD * DIM + 255) / 256, 256, 0, stream>>>(sB_down, B_down, DIM, Vd);

  cast_kernel<<<((NTOK * DIM) + 255) / 256, 256, 0, stream>>>(x, xb, NTOK * DIM);
  router_kernel<<<NTOK, 256, 0, stream>>>(x, W_rt, ebias, w_rout);

  gemm_u_kernel<<<dim3(RPAD / 32, NTOK / 128), 256, 0, stream>>>(xb, DIM, DIM, SAg, P);
  make_u_kernel<<<NTOK, RPAD, 0, stream>>>(P, w_rout, ug);
  gemm_u_kernel<<<dim3(RPAD / 32, NTOK / 128), 256, 0, stream>>>(xb, DIM, DIM, SAu, P);
  make_u_kernel<<<NTOK, RPAD, 0, stream>>>(P, w_rout, uu);

  gateup_kernel<<<dim3(DFF / 64, NTOK / 128), 256, 0, stream>>>(
      xb, W_gate, W_up, ug, uu, Vg, Vu, hidden);

  gemm_u_kernel<<<dim3(RPAD / 32, NTOK / 128), 256, 0, stream>>>(hidden, DFF, DFF, SAd, P);
  make_u_kernel<<<NTOK, RPAD, 0, stream>>>(P, w_rout, ud);

  down_kernel<<<dim3(DIM / 64, NTOK / 128), 256, 0, stream>>>(hidden, W_down, ud, Vd, out);
}
